// LongformerMultiheadAttention_63359357550959
// MI455X (gfx1250) — compile-verified, matrix operand order checked by reading, not run on hardware
//
#include <hip/hip_runtime.h>

#define TSEQ 4096
#define BB   2
#define EE   1024
#define HH   16
#define DD   64
#define WW   256
#define GG   64

typedef __bf16 bf16_t;
typedef __attribute__((ext_vector_type(16))) __bf16 v16bf;
typedef __attribute__((ext_vector_type(8)))  __bf16 v8bf;
typedef __attribute__((ext_vector_type(8)))  float  v8f;

static __device__ __forceinline__ unsigned short bf16_bits(float f) {
  unsigned int u = __float_as_uint(f);
  u += 0x7FFFu + ((u >> 16) & 1u);      // round-to-nearest-even truncation
  return (unsigned short)(u >> 16);
}
static __device__ __forceinline__ bf16_t to_bf16(float f) {
  return __builtin_bit_cast(__bf16, bf16_bits(f));
}
static __device__ __forceinline__ v8f vzero8() {
  v8f z;
#pragma unroll
  for (int i = 0; i < 8; ++i) z[i] = 0.0f;
  return z;
}

// Load one 16(row) x 32(K) bf16 fragment in the CDNA5 WMMA A/B register layout:
// lane L (L<16):  row L,    K = {0..7, 16..23}
// lane L (L>=16): row L-16, K = {8..15, 24..31}
static __device__ __forceinline__ v16bf load_frag(const bf16_t* tile, int ld, int lane) {
  int r  = lane & 15;
  int k0 = (lane & 16) ? 8 : 0;
  const bf16_t* p = tile + (long)r * ld + k0;
  v8bf lo = *(const v8bf*)(p);
  v8bf hi = *(const v8bf*)(p + 16);
  v16bf out;
#pragma unroll
  for (int i = 0; i < 8; ++i) { out[i] = lo[i]; out[i + 8] = hi[i]; }
  return out;
}

static __device__ __forceinline__ v8f wmma_bf16(v16bf a, v16bf b, v8f c) {
  // D = A(16x32) * B(32x16) + C, f32 accumulate
  return __builtin_amdgcn_wmma_f32_16x16x32_bf16(false, a, false, b, (short)0, c,
                                                 false, false);
}

// ---------------------------------------------------------------------------
// Conversion kernels
// ---------------------------------------------------------------------------
__global__ __launch_bounds__(256)
void cvt_x_kernel(const float* __restrict__ q, bf16_t* __restrict__ X,
                  bf16_t* __restrict__ Xg) {
  long idx = ((long)blockIdx.x * 256 + threadIdx.x) * 8;   // over [B,T,E]
  int e = (int)(idx % EE);
  long bt = idx / EE;
  int t = (int)(bt % TSEQ);
  int b = (int)(bt / TSEQ);
  const float* src = q + (((long)t * BB + b) * EE + e);    // input is [T,B,E]
  v8bf pk;
#pragma unroll
  for (int i = 0; i < 8; ++i) pk[i] = to_bf16(src[i]);
  *(v8bf*)(X + ((long)b * TSEQ + t) * EE + e) = pk;
  if (t < GG) *(v8bf*)(Xg + ((long)b * GG + t) * EE + e) = pk;
}

__global__ __launch_bounds__(256)
void cvt_w_kernel(const float* __restrict__ src, bf16_t* __restrict__ dst) {
  long idx = ((long)blockIdx.x * 256 + threadIdx.x) * 8;
  v8bf pk;
#pragma unroll
  for (int i = 0; i < 8; ++i) pk[i] = to_bf16(src[idx + i]);
  *(v8bf*)(dst + idx) = pk;
}

// ---------------------------------------------------------------------------
// GEMM: C[M,N] = A[M,K] * W[N,K]^T (+bias)*scale
// mode 0: bf16 out, row-major [M,N]
// mode 1: bf16 out, transposed to [B,H,D,T]   (for V / Vg)
// mode 2: f32  out, rows m=(b,t) scattered to [T,B,E]  (final output)
// ---------------------------------------------------------------------------
__global__ __launch_bounds__(256)
void gemm_bf16_kernel(const bf16_t* __restrict__ A, const bf16_t* __restrict__ W,
                      const float* __restrict__ bias, void* __restrict__ out,
                      int M, int N, int K, float scale, int mode) {
  (void)M;
  int lane = threadIdx.x & 31;
  int wave = threadIdx.x >> 5;
  int wm = wave & 3;                 // 4 waves along M (32 rows each)
  int wn = wave >> 2;                // 2 waves along N (64 cols each)
  int row0 = blockIdx.y * 128 + wm * 32;
  int col0 = blockIdx.x * 128 + wn * 64;

  v8f c[2][4];
#pragma unroll
  for (int i = 0; i < 2; ++i)
#pragma unroll
    for (int j = 0; j < 4; ++j) c[i][j] = vzero8();

  for (int k = 0; k < K; k += 32) {
    v16bf a0 = load_frag(A + (long)row0 * K + k, K, lane);
    v16bf a1 = load_frag(A + (long)(row0 + 16) * K + k, K, lane);
#pragma unroll
    for (int nt = 0; nt < 4; ++nt) {
      v16bf bfrag = load_frag(W + (long)(col0 + nt * 16) * K + k, K, lane);
      c[0][nt] = wmma_bf16(a0, bfrag, c[0][nt]);
      c[1][nt] = wmma_bf16(a1, bfrag, c[1][nt]);
    }
  }

  int nloc = lane & 15;
  int rbase = (lane & 16) ? 8 : 0;
#pragma unroll
  for (int mt = 0; mt < 2; ++mt) {
#pragma unroll
    for (int nt = 0; nt < 4; ++nt) {
      int n = col0 + nt * 16 + nloc;
      float bv = bias[n];
      int mrow0 = row0 + mt * 16 + rbase;
      if (mode == 0) {
        bf16_t* o = (bf16_t*)out;
#pragma unroll
        for (int r = 0; r < 8; ++r)
          o[(long)(mrow0 + r) * N + n] = to_bf16((c[mt][nt][r] + bv) * scale);
      } else if (mode == 1) {
        bf16_t* o = (bf16_t*)out;           // [B,H,D,T]
        int b = mrow0 / TSEQ;
        int t = mrow0 - b * TSEQ;           // 8 consecutive t per lane
        int h = n >> 6;
        int d = n & 63;
        v8bf pk;
#pragma unroll
        for (int r = 0; r < 8; ++r) pk[r] = to_bf16((c[mt][nt][r] + bv) * scale);
        *(v8bf*)(o + (((long)b * HH + h) * DD + d) * TSEQ + t) = pk;
      } else {
        float* o = (float*)out;             // [T,B,E]
#pragma unroll
        for (int r = 0; r < 8; ++r) {
          int m = mrow0 + r;
          int b = m / TSEQ;
          int t = m - b * TSEQ;
          o[((long)t * BB + b) * (long)N + n] = (c[mt][nt][r] + bv) * scale;
        }
      }
    }
  }
}

// ---------------------------------------------------------------------------
// Flash-style attention. One wave = one 16-query tile of one (b,h).
// mode 0: band queries (all T), keys = global chunk (0..G) + banded window
// mode 1: global-token queries (t<G, q from Qg), keys = all T, no mask
// ---------------------------------------------------------------------------
__global__ __launch_bounds__(256)
void attn_kernel(const bf16_t* __restrict__ Q, const bf16_t* __restrict__ Kc,
                 const bf16_t* __restrict__ Vt, bf16_t* __restrict__ Oattn,
                 int mode) {
  __shared__ unsigned short psh[8][16 * 48];   // per-wave P transpose buffer
  int lane = threadIdx.x & 31;
  int wave = threadIdx.x >> 5;
  int g = blockIdx.x * 8 + wave;
  int ntt = (mode == 0) ? (TSEQ / 16) : (GG / 16);
  int b = g / (HH * ntt);
  int rem = g - b * (HH * ntt);
  int h = rem / ntt;
  int tt = rem - h * ntt;
  int t0 = tt * 16;

  long qrow = (mode == 0) ? ((long)b * TSEQ + t0) : ((long)b * GG + t0);
  const bf16_t* qbase = Q + qrow * EE + h * DD;
  v16bf qa0 = load_frag(qbase, EE, lane);        // d = 0..31
  v16bf qa1 = load_frag(qbase + 32, EE, lane);   // d = 32..63

  const bf16_t* kmat  = Kc + (long)b * TSEQ * EE + h * DD;        // +key*EE
  const bf16_t* vbase = Vt + ((long)b * HH + h) * (long)DD * TSEQ;// +d*T + t

  float mi[8], li[8];
#pragma unroll
  for (int r = 0; r < 8; ++r) { mi[r] = -3.0e38f; li[r] = 0.0f; }
  v8f o[4];
#pragma unroll
  for (int dt = 0; dt < 4; ++dt) o[dt] = vzero8();

  int kb0, kbend;
  if (mode == 0) {
    int lo = (t0 - WW) & ~31;                    // floor-align (sign-safe)
    kb0 = lo < 0 ? 0 : lo;
    int hi = t0 + 15 + WW;
    kbend = hi > (TSEQ - 1) ? (TSEQ - 1) : hi;
  } else { kb0 = 0; kbend = TSEQ - 1; }

  unsigned short* pw = &psh[wave][0];
  int nloc = lane & 15;
  int rbase = (lane & 16) ? 8 : 0;

  for (int phase = 0; phase < 2; ++phase) {
    int cstart, cend; bool use_mask;
    if (phase == 0) {                 // global keys, duplicated in softmax
      if (mode != 0) continue;
      cstart = 0; cend = GG - 1; use_mask = false;
    } else {
      cstart = kb0; cend = kbend; use_mask = (mode == 0);
    }
    for (int kb = cstart; kb <= cend; kb += 32) {
      // ---- scores: S(16q x 32k) = Q(16x64) * K^T ----
      v8f s0 = vzero8(), s1 = vzero8();
      {
        v16bf b00 = load_frag(kmat + (long)kb * EE, EE, lane);
        v16bf b01 = load_frag(kmat + (long)kb * EE + 32, EE, lane);
        s0 = wmma_bf16(qa0, b00, s0);
        s0 = wmma_bf16(qa1, b01, s0);
        v16bf b10 = load_frag(kmat + (long)(kb + 16) * EE, EE, lane);
        v16bf b11 = load_frag(kmat + (long)(kb + 16) * EE + 32, EE, lane);
        s1 = wmma_bf16(qa0, b10, s1);
        s1 = wmma_bf16(qa1, b11, s1);
      }
      if (use_mask) {
        int j0 = kb + nloc;
        int j1 = j0 + 16;
#pragma unroll
        for (int r = 0; r < 8; ++r) {
          int t = t0 + rbase + r;
          if (j0 < t - WW || j0 > t + WW) s0[r] = -1.0e30f;
          if (j1 < t - WW || j1 > t + WW) s1[r] = -1.0e30f;
        }
      }
      // ---- online softmax (row = query, spread over 16 lanes of half) ----
      float alpha[8];
#pragma unroll
      for (int r = 0; r < 8; ++r) {
        float rm = fmaxf(s0[r], s1[r]);
#pragma unroll
        for (int i = 1; i < 16; i <<= 1) rm = fmaxf(rm, __shfl_xor(rm, i, 32));
        float mn = fmaxf(mi[r], rm);
        alpha[r] = __expf(mi[r] - mn);
        mi[r] = mn;
        float p0 = __expf(s0[r] - mn);
        float p1 = __expf(s1[r] - mn);
        s0[r] = p0; s1[r] = p1;
        float rs = p0 + p1;
#pragma unroll
        for (int i = 1; i < 16; i <<= 1) rs += __shfl_xor(rs, i, 32);
        li[r] = li[r] * alpha[r] + rs;
      }
#pragma unroll
      for (int dt = 0; dt < 4; ++dt)
#pragma unroll
        for (int r = 0; r < 8; ++r) o[dt][r] *= alpha[r];
      // ---- rearrange P: C-layout -> A-layout via per-wave LDS ----
#pragma unroll
      for (int r = 0; r < 8; ++r) {
        int m = rbase + r;
        pw[m * 48 + nloc]      = bf16_bits(s0[r]);
        pw[m * 48 + nloc + 16] = bf16_bits(s1[r]);
      }
      v16bf pa = load_frag((const bf16_t*)pw, 48, lane);
      // ---- O += P(16x32) * V(32x64), V transposed -> contiguous keys ----
#pragma unroll
      for (int dt = 0; dt < 4; ++dt) {
        v16bf vb = load_frag(vbase + (long)(dt * 16) * TSEQ + kb, TSEQ, lane);
        o[dt] = wmma_bf16(pa, vb, o[dt]);
      }
    }
  }

  // ---- finalize and store bf16 attn [B,T,E] ----
  long orow0 = (long)b * TSEQ + t0;
#pragma unroll
  for (int dt = 0; dt < 4; ++dt) {
#pragma unroll
    for (int r = 0; r < 8; ++r) {
      float v = o[dt][r] * (1.0f / li[r]);
      int m = rbase + r;
      Oattn[(orow0 + m) * EE + h * DD + dt * 16 + nloc] = to_bf16(v);
    }
  }
}

// ---------------------------------------------------------------------------
extern "C" void kernel_launch(void* const* d_in, const int* in_sizes, int n_in,
                              void* d_out, int out_size, void* d_ws, size_t ws_size,
                              hipStream_t stream) {
  (void)in_sizes; (void)n_in; (void)out_size; (void)ws_size;
  const float* query = (const float*)d_in[0];
  // d_in[1] = attn_mask (structure fixed; unused, as in reference)
  const float* Wq  = (const float*)d_in[2];  const float* bq  = (const float*)d_in[3];
  const float* Wk  = (const float*)d_in[4];  const float* bk  = (const float*)d_in[5];
  const float* Wv  = (const float*)d_in[6];  const float* bv  = (const float*)d_in[7];
  const float* Wqg = (const float*)d_in[8];  const float* bqg = (const float*)d_in[9];
  const float* Wkg = (const float*)d_in[10]; const float* bkg = (const float*)d_in[11];
  const float* Wvg = (const float*)d_in[12]; const float* bvg = (const float*)d_in[13];
  const float* Wo  = (const float*)d_in[14]; const float* bo  = (const float*)d_in[15];

  char* ws = (char*)d_ws;
  size_t off = 0;
  auto take = [&](size_t bytes) -> char* {
    char* p = ws + off; off += (bytes + 255) & ~(size_t)255; return p;
  };
  const size_t BTE = (size_t)BB * TSEQ * EE;
  const size_t WSZ = (size_t)EE * EE;
  bf16_t* Xbf  = (bf16_t*)take(BTE * 2);
  bf16_t* Xg   = (bf16_t*)take((size_t)BB * GG * EE * 2);
  bf16_t* WqB  = (bf16_t*)take(WSZ * 2);
  bf16_t* WkB  = (bf16_t*)take(WSZ * 2);
  bf16_t* WvB  = (bf16_t*)take(WSZ * 2);
  bf16_t* WqgB = (bf16_t*)take(WSZ * 2);
  bf16_t* WkgB = (bf16_t*)take(WSZ * 2);
  bf16_t* WvgB = (bf16_t*)take(WSZ * 2);
  bf16_t* WoB  = (bf16_t*)take(WSZ * 2);
  bf16_t* Qb   = (bf16_t*)take(BTE * 2);
  bf16_t* Kb   = (bf16_t*)take(BTE * 2);
  bf16_t* Vt   = (bf16_t*)take(BTE * 2);   // [B,H,D,T]
  bf16_t* Qg   = (bf16_t*)take((size_t)BB * GG * EE * 2);
  bf16_t* Kg   = (bf16_t*)take(BTE * 2);
  bf16_t* Vgt  = (bf16_t*)take(BTE * 2);   // [B,H,D,T]
  bf16_t* At   = (bf16_t*)take(BTE * 2);

  dim3 blk(256);
  // conversions
  cvt_x_kernel<<<(unsigned)((BTE / 8) / 256), blk, 0, stream>>>(query, Xbf, Xg);
  unsigned wblk = (unsigned)((WSZ / 8) / 256);
  cvt_w_kernel<<<wblk, blk, 0, stream>>>(Wq,  WqB);
  cvt_w_kernel<<<wblk, blk, 0, stream>>>(Wk,  WkB);
  cvt_w_kernel<<<wblk, blk, 0, stream>>>(Wv,  WvB);
  cvt_w_kernel<<<wblk, blk, 0, stream>>>(Wqg, WqgB);
  cvt_w_kernel<<<wblk, blk, 0, stream>>>(Wkg, WkgB);
  cvt_w_kernel<<<wblk, blk, 0, stream>>>(Wvg, WvgB);
  cvt_w_kernel<<<wblk, blk, 0, stream>>>(Wo,  WoB);

  const float scaling = 0.125f;  // D^-0.5 = 64^-0.5
  dim3 gfull(EE / 128, (BB * TSEQ) / 128);   // (8, 64)
  dim3 gq(EE / 128, (BB * GG) / 128);        // (8, 1)
  // projections
  gemm_bf16_kernel<<<gfull, blk, 0, stream>>>(Xbf, WqB,  bq,  Qb,  BB*TSEQ, EE, EE, scaling, 0);
  gemm_bf16_kernel<<<gfull, blk, 0, stream>>>(Xbf, WkB,  bk,  Kb,  BB*TSEQ, EE, EE, 1.0f,    0);
  gemm_bf16_kernel<<<gfull, blk, 0, stream>>>(Xbf, WvB,  bv,  Vt,  BB*TSEQ, EE, EE, 1.0f,    1);
  gemm_bf16_kernel<<<gfull, blk, 0, stream>>>(Xbf, WkgB, bkg, Kg,  BB*TSEQ, EE, EE, 1.0f,    0);
  gemm_bf16_kernel<<<gfull, blk, 0, stream>>>(Xbf, WvgB, bvg, Vgt, BB*TSEQ, EE, EE, 1.0f,    1);
  gemm_bf16_kernel<<<gq,    blk, 0, stream>>>(Xg,  WqgB, bqg, Qg,  BB*GG,   EE, EE, scaling, 0);
  // attention: band+global-key softmax for all queries, then global-token rows
  attn_kernel<<<(BB * HH * (TSEQ / 16)) / 8, blk, 0, stream>>>(Qb, Kb, Vt, At, 0);
  attn_kernel<<<(BB * HH * (GG / 16)) / 8,  blk, 0, stream>>>(Qg, Kg, Vgt, At, 1);
  // output projection, f32 to [T,B,E]
  gemm_bf16_kernel<<<gfull, blk, 0, stream>>>(At, WoB, bo, d_out, BB*TSEQ, EE, EE, 1.0f, 2);
}
